// SparseConvBlock_64785286693647
// MI455X (gfx1250) — matrix operand order repaired and checked by reading, not compiled, not run on hardware
//
#include <hip/hip_runtime.h>

// ---------------------------------------------------------------------------
// SparseConvBlock for MI455X (gfx1250, wave32, WMMA).
//
// Reference: out = relu(batchnorm(einsum('nkc,kcd', gather(feat, nbr), W)))
//   N=100000 voxels, Cin=32, Cout=64, K=27 offsets.
//
// Roofline: 11.1 GFLOP vs ~50MB unique data (fits in 192MB L2) -> compute/
// latency bound; GEMM runs on v_wmma_f32_16x16x32_f16. One wave owns a
// 16-voxel tile; 4 accumulators cover Cout=64; 27*4 = 108 WMMAs per tile
// (fully unrolled -> scheduler hides gather latency behind the matrix pipe).
// W is pre-swizzled to the B-matrix VGPR layout and staged into LDS with
// gfx1250 async global->LDS copies (ASYNCcnt), shared by 8 waves/block.
// BN stats: per-wave reduce + ds_swizzle SWAPX16 cross-half add + global
// f32 atomics; tiny finalize kernel; fused BN+ReLU apply pass.
// ---------------------------------------------------------------------------

#define N_VOX 100000
#define C_IN  32
#define C_OUT 64
#define KOFF  27
#define BN_EPS 1e-5f

typedef __attribute__((ext_vector_type(16))) _Float16 v16h;
typedef __attribute__((ext_vector_type(8)))  float    v8f;

struct U8x32 { uint4 a, b; };   // 32 bytes == v16h

__device__ __forceinline__ v16h pack_v16h(uint4 lo, uint4 hi) {
  U8x32 u; u.a = lo; u.b = hi;
  return __builtin_bit_cast(v16h, u);
}

// --- zero the BN accumulator area (sum[64], sumsq[64], scale[64], shift[64])
__global__ void sc_init_stats(float* __restrict__ stats) {
  if (threadIdx.x < 256) stats[threadIdx.x] = 0.0f;
}

// --- fp32 features -> packed f16 (2 per dword store) ---------------------
__global__ __launch_bounds__(256) void sc_cvt_feat(
    const float* __restrict__ f, unsigned* __restrict__ out) {
  int i = blockIdx.x * blockDim.x + threadIdx.x;
  if (i < (N_VOX * C_IN) / 2) {
    _Float16 ha = (_Float16)f[2 * i];
    _Float16 hb = (_Float16)f[2 * i + 1];
    unsigned ua = __builtin_bit_cast(unsigned short, ha);
    unsigned ub = __builtin_bit_cast(unsigned short, hb);
    out[i] = ua | (ub << 16);
  }
}

// --- W[k][c][n] fp32 -> f16 as B operand (32x16, KxN) in this code's contiguous k order:
//   lane<16 : column n=lane,    VGPR j holds K=2j,2j+1       (K 0..15)
//   lane>=16: column n=lane-16, VGPR j holds K=16+2j,16+2j+1 (K 16..31)
// One 32-byte record per (k, ntile, lane); record = 8 packed-f16 dwords.
__global__ __launch_bounds__(256) void sc_prep_w(
    const float* __restrict__ W, unsigned* __restrict__ w16u) {
  int t = blockIdx.x * blockDim.x + threadIdx.x;
  if (t >= KOFF * 4 * 32 * 8) return;          // 27648 dwords
  int j    = t & 7;
  int rec  = t >> 3;
  int lane = rec & 31;
  int kt   = rec >> 5;                         // k*4 + ntile
  int tn   = kt & 3;
  int k    = kt >> 2;
  int n    = (lane & 15) + tn * 16;
  int kb   = (lane >> 4) * 16;
  int c0   = kb + 2 * j;
  float w0 = W[((size_t)k * C_IN + c0) * C_OUT + n];
  float w1 = W[((size_t)k * C_IN + c0 + 1) * C_OUT + n];
  unsigned ua = __builtin_bit_cast(unsigned short, (_Float16)w0);
  unsigned ub = __builtin_bit_cast(unsigned short, (_Float16)w1);
  w16u[t] = ua | (ub << 16);
}

// --- main gather + WMMA GEMM kernel --------------------------------------
// block = 256 threads = 8 waves; wave w handles voxel tile blockIdx*8+w.
__global__ __launch_bounds__(256) void sc_conv_wmma(
    const _Float16* __restrict__ feat16,      // [N][32] f16
    const uint4*    __restrict__ w16,         // 6912 uint4 (B-layout records)
    const int*      __restrict__ nbr,         // [N][27]
    float*          __restrict__ preout,      // [N][64] pre-BN
    float*          __restrict__ stats)       // sum[64], sumsq[64]
{
  extern __shared__ uint4 ldsW[];             // 6912 uint4 = 110592 B

  // Stage W into LDS with gfx1250 async global->LDS b128 copies (ASYNCcnt).
  // VDST VGPR = LDS byte offset, GV addressing (64-bit VGPR addr, SADDR=off).
  for (int i = threadIdx.x; i < KOFF * 4 * 32 * 2; i += 256) {
    unsigned ldsOff = (unsigned)(size_t)(&ldsW[i]);
    const uint4* gp = w16 + i;
    asm volatile("global_load_async_to_lds_b128 %0, %1, off"
                 :: "v"(ldsOff), "v"(gp)
                 : "memory");
  }
  asm volatile("s_wait_asynccnt 0x0" ::: "memory");
  __syncthreads();

  const int lane    = threadIdx.x & 31;
  const int wave    = threadIdx.x >> 5;
  const int half    = lane >> 4;              // which 16-lane half
  const int m       = lane & 15;
  const int voxBase = (blockIdx.x * 8 + wave) * 16;
  const int vrow    = voxBase + m;            // A-matrix row this lane feeds

  const uint4* f4 = reinterpret_cast<const uint4*>(feat16); // row r: f4[4r..4r+3]

  v8f acc0 = {}, acc1 = {}, acc2 = {}, acc3 = {};

  for (int k = 0; k < KOFF; ++k) {
    int row = -1;
    if (vrow < N_VOX) row = nbr[vrow * KOFF + k];
    // A: 16x32 f16, contiguous k order as in B. lane<16: K0..15 of row m;
    //    lane>=16: K16..31 of row m. Two b128 loads, zero if masked.
    uint4 lo{0u, 0u, 0u, 0u}, hi{0u, 0u, 0u, 0u};
    if (row >= 0) {
      lo = f4[row * 4 + 2 * half];
      hi = f4[row * 4 + 2 * half + 1];
    }
    v16h a = pack_v16h(lo, hi);

    const uint4* wk = ldsW + (size_t)(k * 4) * 64;   // 64 uint4 per ntile
    {
      const uint4* p = wk + lane * 2;
      v16h b = pack_v16h(p[0], p[1]);
      acc0 = __builtin_amdgcn_wmma_f32_16x16x32_f16(false, a, false, b,
                                                    (short)0, acc0, false, false);
    }
    {
      const uint4* p = wk + 64 + lane * 2;
      v16h b = pack_v16h(p[0], p[1]);
      acc1 = __builtin_amdgcn_wmma_f32_16x16x32_f16(false, a, false, b,
                                                    (short)0, acc1, false, false);
    }
    {
      const uint4* p = wk + 128 + lane * 2;
      v16h b = pack_v16h(p[0], p[1]);
      acc2 = __builtin_amdgcn_wmma_f32_16x16x32_f16(false, a, false, b,
                                                    (short)0, acc2, false, false);
    }
    {
      const uint4* p = wk + 192 + lane * 2;
      v16h b = pack_v16h(p[0], p[1]);
      acc3 = __builtin_amdgcn_wmma_f32_16x16x32_f16(false, a, false, b,
                                                    (short)0, acc3, false, false);
    }
  }

  // C/D layout: VGPR r, lanes 0-15 -> (M=r, N=lane); lanes 16-31 -> (M=r+8).
  // Each lane owns column n for 8 rows. Store pre-BN + accumulate BN stats.
  const int ncol  = m;
  const int rbase = voxBase + half * 8;

  auto emit = [&](v8f acc, int t) {
    float s = 0.0f, ss = 0.0f;
#pragma unroll
    for (int r = 0; r < 8; ++r) {
      int grow = rbase + r;
      float v = acc[r];
      if (grow < N_VOX) {
        preout[(size_t)grow * C_OUT + t * 16 + ncol] = v;
        s  += v;
        ss += v * v;
      }
    }
    // cross-half combine (lane <-> lane^16) via ds_swizzle SWAPX16
    s  += __builtin_bit_cast(float,
            __builtin_amdgcn_ds_swizzle(__builtin_bit_cast(int, s), 0x401F));
    ss += __builtin_bit_cast(float,
            __builtin_amdgcn_ds_swizzle(__builtin_bit_cast(int, ss), 0x401F));
    if (half == 0) {
      atomicAdd(&stats[t * 16 + ncol], s);
      atomicAdd(&stats[64 + t * 16 + ncol], ss);
    }
  };
  emit(acc0, 0); emit(acc1, 1); emit(acc2, 2); emit(acc3, 3);
}

// --- fold sums into BN scale/shift ---------------------------------------
__global__ void sc_bn_finalize(const float* __restrict__ gamma,
                               const float* __restrict__ beta,
                               float* __restrict__ stats) {
  int c = threadIdx.x;
  if (c < C_OUT) {
    const float inv_n = 1.0f / (float)N_VOX;
    float mean  = stats[c] * inv_n;
    float var   = stats[64 + c] * inv_n - mean * mean;
    float scale = gamma[c] * rsqrtf(var + BN_EPS);
    stats[128 + c] = scale;
    stats[192 + c] = beta[c] - mean * scale;
  }
}

// --- apply BN + ReLU, float4 vectorized ----------------------------------
__global__ __launch_bounds__(256) void sc_bn_relu(
    const float* __restrict__ preout, const float* __restrict__ stats,
    float* __restrict__ out) {
  int t = blockIdx.x * blockDim.x + threadIdx.x;
  if (t >= (N_VOX * C_OUT) / 4) return;
  const float4* in4 = reinterpret_cast<const float4*>(preout);
  float4*       o4  = reinterpret_cast<float4*>(out);
  float4 x = in4[t];
  int cb = (t * 4) & (C_OUT - 1);
  float4 r;
  r.x = fmaxf(0.0f, x.x * stats[128 + cb + 0] + stats[192 + cb + 0]);
  r.y = fmaxf(0.0f, x.y * stats[128 + cb + 1] + stats[192 + cb + 1]);
  r.z = fmaxf(0.0f, x.z * stats[128 + cb + 2] + stats[192 + cb + 2]);
  r.w = fmaxf(0.0f, x.w * stats[128 + cb + 3] + stats[192 + cb + 3]);
  o4[t] = r;
}

extern "C" void kernel_launch(void* const* d_in, const int* in_sizes, int n_in,
                              void* d_out, int out_size, void* d_ws, size_t ws_size,
                              hipStream_t stream) {
  const float* features = (const float*)d_in[0];   // [N,32] f32
  const int*   nbr      = (const int*)d_in[1];     // [N,27] i32
  const float* W        = (const float*)d_in[2];   // [27,32,64] f32
  const float* gamma    = (const float*)d_in[3];   // [64]
  const float* beta     = (const float*)d_in[4];   // [64]

  // Workspace layout (16B aligned throughout):
  //   preout : N*64 f32                 (25.6 MB)
  //   stats  : 256 f32 (sum|sumsq|scale|shift)
  //   feat16 : N*32 f16                 (6.4 MB)
  //   w16    : 27648 dwords B-layout W  (110.6 KB)
  float*    preout = (float*)d_ws;
  float*    stats  = preout + (size_t)N_VOX * C_OUT;
  _Float16* feat16 = (_Float16*)(stats + 256);
  unsigned* w16u   = (unsigned*)((char*)feat16 + (size_t)N_VOX * C_IN * 2);

  sc_init_stats<<<1, 256, 0, stream>>>(stats);

  sc_cvt_feat<<<((N_VOX * C_IN / 2) + 255) / 256, 256, 0, stream>>>(
      features, (unsigned*)feat16);

  sc_prep_w<<<(KOFF * 4 * 32 * 8 + 255) / 256, 256, 0, stream>>>(W, w16u);

  const int tiles  = (N_VOX + 15) / 16;      // 6250
  const int blocks = (tiles + 7) / 8;        // 782 blocks x 8 waves
  const size_t ldsBytes = (size_t)KOFF * 4 * 32 * 32;  // 110592 B
  sc_conv_wmma<<<blocks, 256, ldsBytes, stream>>>(
      feat16, (const uint4*)w16u, nbr, preout, stats);

  sc_bn_finalize<<<1, 64, 0, stream>>>(gamma, beta, stats);

  sc_bn_relu<<<((N_VOX * C_OUT / 4) + 255) / 256, 256, 0, stream>>>(
      preout, stats, (float*)d_out);
}
